// BertOutput_61813169324802
// MI455X (gfx1250) — 8-bit matrix operand order repaired and checked by reading, compiled, not run on hardware
//
#include <hip/hip_runtime.h>

typedef __attribute__((ext_vector_type(8))) int v8i;

#define HID   1024
#define INTER 4096
#define NTOK  16384
#define CLIPV 2.5f
#define KT    (INTER / 64)
#define MT_WG 2

__global__ void k_init(unsigned* slots) {
    if (threadIdx.x < 2) slots[threadIdx.x] = 0u;
}

__global__ void k_maxabs(const float* __restrict__ x, long long n, unsigned* slot) {
    long long i      = (long long)blockIdx.x * blockDim.x + threadIdx.x;
    long long stride = (long long)gridDim.x * blockDim.x;
    float m = 0.0f;
    for (; i < n; i += stride) m = fmaxf(m, fabsf(x[i]));
#pragma unroll
    for (int off = 16; off > 0; off >>= 1) m = fmaxf(m, __shfl_xor(m, off, 32));
    if ((threadIdx.x & 31) == 0) atomicMax(slot, __float_as_uint(m));
}

__device__ __forceinline__ int quant1(float v, float s) {
    return (int)rintf(fminf(fmaxf(v, -CLIPV), CLIPV) * s);
}

__global__ void k_quantW(const float* __restrict__ W, const unsigned* __restrict__ slots,
                         signed char* __restrict__ Wq) {
    int tid = blockIdx.x * blockDim.x + threadIdx.x;
    int q   = tid & (INTER / 4 - 1);
    int n   = tid >> 10;
    float m = fminf(__uint_as_float(slots[0]), CLIPV);
    float s = 127.0f / m;
    float4 xv = *(const float4*)(W + (size_t)n * INTER + (size_t)q * 4);
    int b0 = quant1(xv.x, s), b1 = quant1(xv.y, s), b2 = quant1(xv.z, s), b3 = quant1(xv.w, s);
    unsigned pack = (b0 & 255) | ((b1 & 255) << 8) | ((b2 & 255) << 16) | ((b3 & 255) << 24);
    int kt = q >> 4;
    int ql = q & 15;
    int j  = ((ql >> 3) << 2) | (ql & 3);
    int h  = (ql >> 2) & 1;
    int L  = h * 16 + (n & 15);
    size_t blk = ((size_t)(n >> 4) * KT + kt) * 1024;
    *(unsigned*)(Wq + blk + L * 32 + j * 4) = pack;
}

__global__ void k_quantX(const float* __restrict__ X, const unsigned* __restrict__ slots,
                         signed char* __restrict__ Aq) {
    int tid = blockIdx.x * blockDim.x + threadIdx.x;
    int q   = tid & (INTER / 4 - 1);
    int t   = tid >> 10;
    float m = fminf(__uint_as_float(slots[1]), CLIPV);
    float s = 127.0f / m;
    float4 xv = *(const float4*)(X + (size_t)t * INTER + (size_t)q * 4);
    int b0 = quant1(xv.x, s), b1 = quant1(xv.y, s), b2 = quant1(xv.z, s), b3 = quant1(xv.w, s);
    unsigned pack = (b0 & 255) | ((b1 & 255) << 8) | ((b2 & 255) << 16) | ((b3 & 255) << 24);
    int kt = q >> 4;
    int ql = q & 15;
    int j  = ((ql >> 3) << 2) | (ql & 3);
    int h  = (ql >> 2) & 1;
    int L  = h * 16 + (t & 15);
    size_t blk = ((size_t)(t >> 4) * KT + kt) * 1024;
    *(unsigned*)(Aq + blk + L * 32 + j * 4) = pack;
}

__device__ __forceinline__ v8i ld8(const int* __restrict__ p) {
    int4 lo = *(const int4*)(p);
    int4 hi = *(const int4*)(p + 4);
    v8i r;
    r[0] = lo.x; r[1] = lo.y; r[2] = lo.z; r[3] = lo.w;
    r[4] = hi.x; r[5] = hi.y; r[6] = hi.z; r[7] = hi.w;
    return r;
}

__global__ __launch_bounds__(512, 1) void k_gemm_ln(
    const signed char* __restrict__ Aq, const signed char* __restrict__ Bq,
    const unsigned* __restrict__ slots,
    const float* __restrict__ bias, const float* __restrict__ resid,
    const float* __restrict__ gamma, const float* __restrict__ beta,
    float* __restrict__ out) {
    const int lane = threadIdx.x & 31;
    const int w    = threadIdx.x >> 5;
    const int mt   = blockIdx.x;

    v8i acc[MT_WG][4] = {};

    const int* __restrict__ Ab =
        (const int*)(Aq + (size_t)mt * MT_WG * KT * 1024) + lane * 8;
    const int* __restrict__ Bb =
        (const int*)(Bq + (size_t)(w * 4) * KT * 1024) + lane * 8;

    v8i abuf[2][MT_WG];
    v8i bbuf[2][4];
#pragma unroll
    for (int m = 0; m < MT_WG; ++m) abuf[0][m] = ld8(Ab + m * KT * 256);
#pragma unroll
    for (int i = 0; i < 4; ++i) bbuf[0][i] = ld8(Bb + i * KT * 256);
    __builtin_amdgcn_sched_barrier(0);

#pragma unroll
    for (int kt = 0; kt < KT; ++kt) {
        const int cur = kt & 1, nxt = cur ^ 1;
        if (kt + 1 < KT) {
#pragma unroll
            for (int m = 0; m < MT_WG; ++m)
                abuf[nxt][m] = ld8(Ab + (m * KT + kt + 1) * 256);
#pragma unroll
            for (int i = 0; i < 4; ++i)
                bbuf[nxt][i] = ld8(Bb + (i * KT + kt + 1) * 256);
        }
        __builtin_amdgcn_sched_barrier(0);
#pragma unroll
        for (int i = 0; i < 4; ++i) {
#pragma unroll
            for (int m = 0; m < MT_WG; ++m)
                acc[m][i] = __builtin_amdgcn_wmma_i32_16x16x64_iu8(
                    true, abuf[cur][m], true, bbuf[cur][i], acc[m][i],
                    false, false);
        }
        __builtin_amdgcn_sched_barrier(0);
    }

    const float mW    = fminf(__uint_as_float(slots[0]), CLIPV);
    const float mX    = fminf(__uint_as_float(slots[1]), CLIPV);
    const float scale = (mW * mX) / (127.0f * 127.0f);
    const int half = lane >> 4, nl = lane & 15;
    const int t0   = mt * (MT_WG * 16);

    float yv[MT_WG][4][8];
#pragma unroll
    for (int m = 0; m < MT_WG; ++m) {
#pragma unroll
        for (int i = 0; i < 4; ++i) {
            int col = (w * 4 + i) * 16 + nl;
            float bcol = bias[col];
#pragma unroll
            for (int d = 0; d < 8; ++d) {
                int row = m * 16 + d + half * 8;
                yv[m][i][d] = (float)acc[m][i][d] * scale + bcol +
                              resid[(size_t)(t0 + row) * HID + col];
            }
        }
    }

    __shared__ float lsum[MT_WG * 16], lsq[MT_WG * 16];
    if (threadIdx.x < MT_WG * 16) { lsum[threadIdx.x] = 0.0f; lsq[threadIdx.x] = 0.0f; }
    __syncthreads();

#pragma unroll
    for (int m = 0; m < MT_WG; ++m) {
#pragma unroll
        for (int d = 0; d < 8; ++d) {
            float s1 = 0.0f, s2 = 0.0f;
#pragma unroll
            for (int i = 0; i < 4; ++i) { float v = yv[m][i][d]; s1 += v; s2 += v * v; }
#pragma unroll
            for (int msk = 1; msk < 16; msk <<= 1) {
                s1 += __shfl_xor(s1, msk, 32);
                s2 += __shfl_xor(s2, msk, 32);
            }
            if (nl == 0) {
                int row = m * 16 + d + half * 8;
                atomicAdd(&lsum[row], s1);
                atomicAdd(&lsq[row], s2);
            }
        }
    }
    __syncthreads();

#pragma unroll
    for (int m = 0; m < MT_WG; ++m) {
#pragma unroll
        for (int i = 0; i < 4; ++i) {
            int col = (w * 4 + i) * 16 + nl;
            float g = gamma[col], be = beta[col];
#pragma unroll
            for (int d = 0; d < 8; ++d) {
                int row  = m * 16 + d + half * 8;
                float mean = lsum[row] * (1.0f / (float)HID);
                float var  = lsq[row] * (1.0f / (float)HID) - mean * mean;
                float rstd = rsqrtf(var + 1e-12f);
                out[(size_t)(t0 + row) * HID + col] =
                    g * (yv[m][i][d] - mean) * rstd + be;
            }
        }
    }
}

extern "C" void kernel_launch(void* const* d_in, const int* in_sizes, int n_in,
                              void* d_out, int out_size, void* d_ws, size_t ws_size,
                              hipStream_t stream) {
    const float* x     = (const float*)d_in[0];
    const float* resid = (const float*)d_in[1];
    const float* W     = (const float*)d_in[2];
    const float* bias  = (const float*)d_in[3];
    const float* gamma = (const float*)d_in[4];
    const float* beta  = (const float*)d_in[5];
    float* out = (float*)d_out;

    unsigned*    slots = (unsigned*)d_ws;
    signed char* Wq    = (signed char*)d_ws + 1024;
    signed char* Aq    = Wq + (size_t)HID * INTER;

    k_init<<<1, 32, 0, stream>>>(slots);
    k_maxabs<<<2048, 256, 0, stream>>>(W, (long long)HID * INTER, slots + 0);
    k_maxabs<<<8192, 256, 0, stream>>>(x, (long long)NTOK * INTER, slots + 1);
    k_quantW<<<(HID * INTER / 4) / 256, 256, 0, stream>>>(W, slots, Wq);
    k_quantX<<<(NTOK * INTER / 4) / 256, 256, 0, stream>>>(x, slots, Aq);
    k_gemm_ln<<<NTOK / (MT_WG * 16), 512, 0, stream>>>(Aq, Wq, slots, bias, resid,
                                                       gamma, beta, out);
}
